// MultiHeadGraphAttentionLayer_54554674593975
// MI455X (gfx1250) — matrix operand order repaired and checked by reading, not compiled, not run on hardware
//
#include <hip/hip_runtime.h>
#include <hip/hip_bf16.h>
#include <math.h>

#define NN   6144
#define DD   65
#define HH   4
#define KP   96      // K=65 padded to 3x32 for wmma_f32_16x16x32_f16
#define EPSF 1e-7f
#define MINF 1e-15f

typedef _Float16 v16h __attribute__((ext_vector_type(16)));
typedef _Float16 v8h  __attribute__((ext_vector_type(8)));
typedef float    v8f  __attribute__((ext_vector_type(8)));

union V16 { v16h v; v8h h[2]; };

__device__ __forceinline__ float arcoshf_(float z) {
  return logf(z + sqrtf(fmaxf(z * z - 1.0f, 0.0f)));
}

// Load a 16-element f16 fragment made of two contiguous 8-half runs.
__device__ __forceinline__ v16h ld2x8(const _Float16* p0, const _Float16* p1) {
  V16 u;
  u.h[0] = *(const v8h*)p0;
  u.h[1] = *(const v8h*)p1;
  return u.v;
}

// ---------------------------------------------------------------------------
// Kernel 1: per (node, head) compute att/red = hyp_linear(x, W) and emit
//   attA (f16, K padded to 96), attB (tail negated so theta is a plain GEMM),
//   PvT  (f16, TRANSPOSED [h][d][node] for contiguous staging), lamm1 (f32).
// ---------------------------------------------------------------------------
__global__ void __launch_bounds__(64) prep_kernel(
    const float* __restrict__ x, const float* __restrict__ W_att,
    const float* __restrict__ W_data,
    _Float16* __restrict__ attA, _Float16* __restrict__ attB,
    _Float16* __restrict__ PvT, float* __restrict__ lamm1)
{
  int id = blockIdx.x * 64 + threadIdx.x;
  if (id >= NN * HH) return;
  int node = id >> 2;
  int h    = id & 3;

  const float* xr = x + (size_t)node * DD;
  float y[64];
  float n2 = 0.f;
  #pragma unroll
  for (int k = 0; k < 64; ++k) { float v = xr[k + 1]; y[k] = v; n2 += v * v; }
  float nn  = fmaxf(sqrtf(n2), MINF);
  float lsc = arcoshf_(fmaxf(xr[0], 1.0f + EPSF)) / nn;   // logmap0 scale

  // ---- att = hyp_linear(x, W_att[h]) ----
  {
    const float* W = W_att + (size_t)h * DD * DD;
    float nz2 = 0.f;
    for (int jj = 1; jj < 65; ++jj) {
      float acc = 0.f;
      #pragma unroll
      for (int k = 0; k < 64; ++k) acc += y[k] * W[jj * DD + k + 1];
      float zv = lsc * acc;
      nz2 += zv * zv;
    }
    float nz = fmaxf(sqrtf(nz2), MINF);
    float sh = sinhf(nz) / nz;
    float a0 = sqrtf(1.0f + sh * sh * nz2);   // hyp_proj time component
    _Float16* Ar = attA + ((size_t)h * NN + node) * KP;
    _Float16* Br = attB + ((size_t)h * NN + node) * KP;
    Ar[0] = (_Float16)a0;  Br[0] = (_Float16)a0;
    float c = sh * lsc;
    for (int jj = 1; jj < 65; ++jj) {
      float acc = 0.f;
      #pragma unroll
      for (int k = 0; k < 64; ++k) acc += y[k] * W[jj * DD + k + 1];
      float av = c * acc;
      Ar[jj] = (_Float16)av;  Br[jj] = (_Float16)(-av);  // Minkowski sign folded
    }
    for (int k = 65; k < KP; ++k) { Ar[k] = (_Float16)0.f; Br[k] = (_Float16)0.f; }
  }

  // ---- red = hyp_linear(x, W_data[h]) -> poincare pv, lam ----
  {
    const float* W = W_data + (size_t)h * DD * DD;
    float nz2 = 0.f;
    for (int jj = 1; jj < 65; ++jj) {
      float acc = 0.f;
      #pragma unroll
      for (int k = 0; k < 64; ++k) acc += y[k] * W[jj * DD + k + 1];
      float zv = lsc * acc;
      nz2 += zv * zv;
    }
    float nz   = fmaxf(sqrtf(nz2), MINF);
    float sh   = sinhf(nz) / nz;
    float r0   = sqrtf(1.0f + sh * sh * nz2);
    float inv0 = 1.0f / (r0 + 1.0f);
    float g    = sh * inv0;               // pv[d] = g * z[d+1]
    float sp   = g * g * nz2;             // ||pv||^2 analytically
    float lam  = 2.0f / fmaxf(1.0f - sp, MINF);
    float c    = lam * g * lsc;
    for (int d = 0; d < 64; ++d) {
      float acc = 0.f;
      #pragma unroll
      for (int k = 0; k < 64; ++k) acc += y[k] * W[(d + 1) * DD + k + 1];
      PvT[((size_t)h * 64 + d) * NN + node] = (_Float16)(c * acc);
    }
    lamm1[(size_t)h * NN + node] = lam - 1.0f;
  }
}

// ---------------------------------------------------------------------------
// Kernel 1b: W_lin -> transposed f16  WlT[272][256], WlT[col][k]=Wl[col][k+1]
// ---------------------------------------------------------------------------
__global__ void __launch_bounds__(256) wlconv_kernel(
    const float* __restrict__ Wl, _Float16* __restrict__ WlT)
{
  int idx = blockIdx.x * 256 + threadIdx.x;      // 272*256 total
  int col = idx >> 8, k = idx & 255;
  WlT[idx] = (col < 260) ? (_Float16)Wl[(size_t)col * 257 + k + 1]
                         : (_Float16)0.f;
}

// ---------------------------------------------------------------------------
// Kernel 2 (WMMA): flash-style fused attention per head, software-pipelined.
// 128-thread block (4 waves) owns a 64-row i-tile.  Double-buffered LDS
// staging: chunk k+1's global loads (32-bit byte offsets off kernel-arg
// bases -> global addressing) are issued at the top of chunk k and land
// while theta WMMAs + gate run; ds_stores into the other buffer go right
// before the end-of-chunk barrier.  2 barriers/chunk.
// ---------------------------------------------------------------------------
__global__ void __launch_bounds__(128) head_kernel(
    const _Float16* __restrict__ attA, const _Float16* __restrict__ attB,
    const _Float16* __restrict__ PvT,  const float* __restrict__ lamm1,
    const float* __restrict__ adj, const float* __restrict__ W_out,
    float* __restrict__ hsb)
{
  const int tid  = threadIdx.x;
  const int lane = tid & 31;
  const int wv   = tid >> 5;
  const int i0   = blockIdx.x * 64;
  const int iw   = i0 + wv * 16;
  const int h    = blockIdx.y;
  const int m    = lane & 15;
  const int hi   = lane >> 4;

  const _Float16* A  = attA + (size_t)h * NN * KP;
  const float*    Wo = W_out + (size_t)h * DD * DD;

  __shared__ float    adjT[2][32][68];    // [buf][jl][i-local 0..63]
  __shared__ _Float16 PvTl[2][64][40];    // [buf][d][jl]
  __shared__ _Float16 Bs[2][32][104];     // [buf][jl][k]
  __shared__ float    lam_s[2][32];
  __shared__ _Float16 WoT[80][64];        // WoT[n][k] = Wo[n][k+1]
  __shared__ _Float16 coefH[4][16][40];   // gated coef, f16 (A-frag friendly)
  __shared__ float    numT[4][16][68];
  __shared__ float    rowN[4][16], rowD[4][16];
  __shared__ float    betaS[4][16];
  __shared__ float    zT[4][16][84];

  // stage WoT once per block (used only in epilogue; any barrier covers it)
  for (int g = tid; g < 80 * 64; g += 128) {
    int col = g >> 6, k = g & 63;
    WoT[col][k] = (col < 65) ? (_Float16)Wo[col * DD + k + 1] : (_Float16)0.f;
  }

  // ---- 32-bit byte offsets (bases stay the kernel args -> global loads) ----
  unsigned aOff[4]; float4* aDst[4];
  #pragma unroll
  for (int it = 0; it < 4; ++it) {        // adj: 32 x 64 f32 = 512 f4
    int g = tid + it * 128, row = g >> 4, c4 = g & 15;
    aDst[it] = (float4*)&adjT[0][row][c4 * 4];
    aOff[it] = (unsigned)((row * NN + i0 + c4 * 4) * 4);
  }
  unsigned bOff[3]; float4* bDst[3];
  #pragma unroll
  for (int it = 0; it < 3; ++it) {        // attB: 32 x 96 f16 = 384 f4
    int g = tid + it * 128, row = g / 12, c = g - row * 12;
    bDst[it] = (float4*)&Bs[0][row][c * 8];
    bOff[it] = (unsigned)(((size_t)h * NN * KP + row * KP + c * 8) * 2);
  }
  unsigned pOff[2]; float4* pDst[2];
  #pragma unroll
  for (int it = 0; it < 2; ++it) {        // PvT: 64 x 32 f16 = 256 f4
    int g = tid + it * 128, row = g >> 2, c = g & 3;
    pDst[it] = (float4*)&PvTl[0][row][c * 8];
    pOff[it] = (unsigned)(((size_t)h * 64 * NN + row * NN + c * 8) * 2);
  }
  unsigned lOff = (unsigned)(((size_t)h * NN + tid) * 4);

  float4 rA[4], rB[3], rP[2]; float rL = 0.f;
  auto LOAD = [&]() {
    #pragma unroll
    for (int it = 0; it < 4; ++it) {
      rA[it] = *(const float4*)((const char*)adj + aOff[it]);
      aOff[it] += 32u * NN * 4u;
    }
    #pragma unroll
    for (int it = 0; it < 3; ++it) {
      rB[it] = *(const float4*)((const char*)attB + bOff[it]);
      bOff[it] += 32u * KP * 2u;
    }
    #pragma unroll
    for (int it = 0; it < 2; ++it) {
      rP[it] = *(const float4*)((const char*)PvT + pOff[it]);
      pOff[it] += 64u;
    }
    if (tid < 32) rL = *(const float*)((const char*)lamm1 + lOff);
    lOff += 128u;
  };
  auto STORE = [&](int buf) {
    #pragma unroll
    for (int it = 0; it < 4; ++it)
      *(float4*)((char*)aDst[it] + buf * (int)sizeof(adjT[0])) = rA[it];
    #pragma unroll
    for (int it = 0; it < 3; ++it)
      *(float4*)((char*)bDst[it] + buf * (int)sizeof(Bs[0])) = rB[it];
    #pragma unroll
    for (int it = 0; it < 2; ++it)
      *(float4*)((char*)pDst[it] + buf * (int)sizeof(PvTl[0])) = rP[it];
    if (tid < 32) lam_s[buf][tid] = rL;
  };

  // A fragments: rows iw..iw+15, two contiguous 8-half runs per K-step
  v16h af[3];
  #pragma unroll
  for (int ks = 0; ks < 3; ++ks) {
    const _Float16* pA = A + (size_t)(iw + m) * KP + ks * 32 + hi * 8;
    af[ks] = ld2x8(pA, pA + 16);
  }

  v8f acc[4] = {};
  float rr[8] = {}, dd[8] = {};   // per-row (r + hi*8) norm^2 / den partials

  LOAD();        // chunk 0
  STORE(0);
  __syncthreads();

  for (int j0 = 0; j0 < NN; j0 += 32) {
    const int  cur  = (j0 >> 5) & 1;
    const bool more = (j0 + 32) < NN;
    if (more) LOAD();               // chunk k+1 in flight during compute

    // ---- theta = attA @ attB^T ----
    v8f th[2] = {};
    #pragma unroll
    for (int ks = 0; ks < 3; ++ks) {
      const _Float16* pb0 = &Bs[cur][m][ks * 32 + hi * 8];
      const _Float16* pb1 = &Bs[cur][16 + m][ks * 32 + hi * 8];
      v16h b0 = ld2x8(pb0, pb0 + 16);
      v16h b1 = ld2x8(pb1, pb1 + 16);
      th[0] = __builtin_amdgcn_wmma_f32_16x16x32_f16(false, af[ks], false, b0,
                                                     (short)0, th[0], false, false);
      th[1] = __builtin_amdgcn_wmma_f32_16x16x32_f16(false, af[ks], false, b1,
                                                     (short)0, th[1], false, false);
    }

    // ---- gate + in-register row accumulation ----
    float lamv0 = lam_s[cur][m], lamv1 = lam_s[cur][16 + m];
    #pragma unroll
    for (int sub = 0; sub < 2; ++sub) {
      float lamv = sub ? lamv1 : lamv0;
      #pragma unroll
      for (int r = 0; r < 8; ++r) {
        int il = r + hi * 8;
        int jl = sub * 16 + m;
        float adjv = adjT[cur][jl][wv * 16 + il];   // adj_t[i,j]
        float c = 0.f;
        if (adjv > 0.01f) {
          float t  = fmaxf(th[sub][r], 1.0f + EPSF);
          float ac = arcoshf_(t);
          c = -adjv * fminf(ac * ac, 50.0f);
        }
        coefH[wv][il][jl] = (_Float16)c;
        rr[r] += c * c;
        dd[r] += fabsf(c) * lamv;
      }
    }
    __syncthreads();                // coef transpose visibility

    // ---- num += coef @ Pv ----
    {
      const _Float16* pc = &coefH[wv][m][hi * 8];
      v16h cf = ld2x8(pc, pc + 16);
      #pragma unroll
      for (int dt = 0; dt < 4; ++dt) {
        const _Float16* pp = &PvTl[cur][dt * 16 + m][hi * 8];
        v16h pf = ld2x8(pp, pp + 16);
        acc[dt] = __builtin_amdgcn_wmma_f32_16x16x32_f16(false, cf, false, pf,
                                                         (short)0, acc[dt], false, false);
      }
    }
    if (more) STORE(cur ^ 1);       // loads have landed; fill other buffer
    __syncthreads();                // end of chunk
  }

  // shfl-tree reduce row partials across the 16 n-lanes of each half
  #pragma unroll
  for (int o = 1; o < 16; o <<= 1)
    #pragma unroll
    for (int r = 0; r < 8; ++r) {
      rr[r] += __shfl_xor(rr[r], o);
      dd[r] += __shfl_xor(dd[r], o);
    }
  if (m == 0)
    #pragma unroll
    for (int r = 0; r < 8; ++r) {
      rowN[wv][r + hi * 8] = rr[r];
      rowD[wv][r + hi * 8] = dd[r];
    }

  // dump raw num tile
  #pragma unroll
  for (int dt = 0; dt < 4; ++dt)
    #pragma unroll
    for (int r = 0; r < 8; ++r)
      numT[wv][r + hi * 8][dt * 16 + m] = acc[dt][r];
  __syncthreads();

  // per-row scalar chain folds into a single scale beta
  if (lane < 16) {
    const int row = lane;
    float raw2 = 0.f;
    for (int d = 0; d < 64; ++d) { float v = numT[wv][row][d]; raw2 += v * v; }
    float norm = fmaxf(sqrtf(rowN[wv][row]), 1e-12f);
    float den  = fmaxf(rowD[wv][row] / norm, MINF);
    float inv  = 1.0f / (norm * den);
    float s2   = raw2 * inv * inv;
    float nv   = fmaxf(sqrtf(s2), MINF);
    float msc  = tanhf(0.5f * atanhf(fminf(nv, 1.0f - EPSF))) / nv;
    float sm   = msc * msc * s2;
    float dH   = fmaxf(1.0f - sm, MINF);
    float hb0  = (1.0f + sm) / dH;
    float yn2  = 4.0f * sm / (dH * dH);
    float nnv  = fmaxf(sqrtf(yn2), MINF);
    float lsc  = arcoshf_(fmaxf(hb0, 1.0f + EPSF)) / nnv;
    betaS[wv][row] = lsc * (2.0f * msc * inv / dH);
  }
  __syncthreads();

  // ---- z = (beta*num) @ WoT^T : 16x64x80 WMMA GEMM ----
  v16h caf[2];
  {
    float b = betaS[wv][m];
    #pragma unroll
    for (int ks = 0; ks < 2; ++ks)
      #pragma unroll
      for (int e = 0; e < 16; ++e) {
        int k = ks * 32 + ((e < 8) ? e : e + 8) + hi * 8;
        caf[ks][e] = (_Float16)(numT[wv][m][k] * b);
      }
  }
  v8f zacc[5] = {};
  #pragma unroll
  for (int nt = 0; nt < 5; ++nt)
    #pragma unroll
    for (int ks = 0; ks < 2; ++ks) {
      const _Float16* pw = &WoT[nt * 16 + m][ks * 32 + hi * 8];
      v16h bf = ld2x8(pw, pw + 16);
      zacc[nt] = __builtin_amdgcn_wmma_f32_16x16x32_f16(false, caf[ks], false, bf,
                                                        (short)0, zacc[nt], false, false);
    }
  #pragma unroll
  for (int nt = 0; nt < 5; ++nt)
    #pragma unroll
    for (int r = 0; r < 8; ++r)
      zT[wv][r + hi * 8][nt * 16 + m] = zacc[nt][r];
  __syncthreads();

  if (lane < 16) {
    const int row = lane;
    float nz2 = 0.f;
    for (int jj = 1; jj < 65; ++jj) { float zv = zT[wv][row][jj]; nz2 += zv * zv; }
    float nz = fmaxf(sqrtf(nz2), MINF);
    float sh = sinhf(nz) / nz;
    float* o = hsb + ((size_t)(iw + row) * HH + h) * DD;
    o[0] = sqrtf(1.0f + sh * sh * nz2);
    for (int jj = 1; jj < 65; ++jj) o[jj] = sh * zT[wv][row][jj];
  }
}

// ---------------------------------------------------------------------------
// Kernel 3 (WMMA): concat heads -> to_poincare -> to_hyperboloid ->
// hyp_linear(W_lin).  One wave per 16 nodes: per-node scale lsc*2/dH folds
// into the f16 hb fragment; z = hbT @ WlT^T is a 16x256x272 WMMA GEMM.
// ---------------------------------------------------------------------------
__global__ void __launch_bounds__(32) final_kernel(
    const float* __restrict__ hsb, const _Float16* __restrict__ WlT,
    float* __restrict__ out)
{
  const int lane  = threadIdx.x;
  const int node0 = blockIdx.x * 16;
  const int m     = lane & 15;
  const int hi    = lane >> 4;

  __shared__ _Float16 hbT[16][264];   // scaled hb tail (f16), stride 528B
  __shared__ float    zS[16][276];    // p scratch, then z tile (16 x 272)

  if (lane < 16) {
    int node = node0 + lane;
    const float* hr = hsb + (size_t)node * 260;   // (node*H+hh)*65 contiguous
    float s = 0.f;
    #pragma unroll
    for (int hh = 0; hh < HH; ++hh) {
      float t0 = hr[hh * 65] + 1.0f;
      for (int d = 0; d < 64; ++d) {
        float p = hr[hh * 65 + 1 + d] / t0;       // to_poincare
        zS[lane][hh * 64 + d] = p;
        s += p * p;
      }
    }
    float dH  = fmaxf(1.0f - s, MINF);
    float hb0 = (1.0f + s) / dH;
    float nn  = fmaxf(2.0f * sqrtf(s) / dH, MINF);
    float lsc = arcoshf_(fmaxf(hb0, 1.0f + EPSF)) / nn;
    float cf2 = lsc * 2.0f / dH;
    for (int k = 0; k < 256; ++k) hbT[lane][k] = (_Float16)(cf2 * zS[lane][k]);
    for (int k = 256; k < 264; ++k) hbT[lane][k] = (_Float16)0.f;
  }
  __syncthreads();

  v16h af[8];
  #pragma unroll
  for (int ks = 0; ks < 8; ++ks) {
    const _Float16* pA = &hbT[m][ks * 32 + hi * 8];
    af[ks] = ld2x8(pA, pA + 16);
  }
  #pragma unroll
  for (int nt = 0; nt < 17; ++nt) {
    v8f zac = {};
    #pragma unroll
    for (int ks = 0; ks < 8; ++ks) {
      const _Float16* pw = WlT + (size_t)(nt * 16 + m) * 256 + ks * 32 + hi * 8;
      v16h bf = ld2x8(pw, pw + 16);
      zac = __builtin_amdgcn_wmma_f32_16x16x32_f16(false, af[ks], false, bf,
                                                   (short)0, zac, false, false);
    }
    #pragma unroll
    for (int r = 0; r < 8; ++r)
      zS[r + hi * 8][nt * 16 + m] = zac[r];
  }
  __syncthreads();

  if (lane < 16) {
    int node = node0 + lane;
    float nz2 = 0.f;
    for (int jj = 1; jj < 260; ++jj) { float zv = zS[lane][jj]; nz2 += zv * zv; }
    float nz = fmaxf(sqrtf(nz2), MINF);
    float sh = sinhf(nz) / nz;
    float* o = out + (size_t)node * 260;
    o[0] = sqrtf(1.0f + sh * sh * nz2);
    for (int jj = 1; jj < 260; ++jj) o[jj] = sh * zS[lane][jj];
  }
}

// ---------------------------------------------------------------------------
extern "C" void kernel_launch(void* const* d_in, const int* in_sizes, int n_in,
                              void* d_out, int out_size, void* d_ws, size_t ws_size,
                              hipStream_t stream)
{
  (void)in_sizes; (void)n_in; (void)out_size; (void)ws_size;
  const float* x      = (const float*)d_in[0];
  const float* adj    = (const float*)d_in[1];
  const float* W_att  = (const float*)d_in[2];
  const float* W_data = (const float*)d_in[3];
  const float* W_out  = (const float*)d_in[4];
  const float* W_lin  = (const float*)d_in[5];

  char* ws = (char*)d_ws;
  size_t off = 0;
  _Float16* attA = (_Float16*)(ws + off); off += (size_t)HH * NN * KP * sizeof(_Float16);
  _Float16* attB = (_Float16*)(ws + off); off += (size_t)HH * NN * KP * sizeof(_Float16);
  _Float16* PvT  = (_Float16*)(ws + off); off += (size_t)HH * 64 * NN * sizeof(_Float16);
  float* lamm1   = (float*)(ws + off);    off += (size_t)HH * NN * sizeof(float);
  float* hsb     = (float*)(ws + off);    off += (size_t)NN * HH * DD * sizeof(float);
  _Float16* WlT  = (_Float16*)(ws + off); off += (size_t)272 * 256 * sizeof(_Float16);

  prep_kernel<<<(NN * HH + 63) / 64, 64, 0, stream>>>(x, W_att, W_data,
                                                      attA, attB, PvT, lamm1);
  wlconv_kernel<<<272, 256, 0, stream>>>(W_lin, WlT);
  head_kernel<<<dim3(NN / 64, HH), 128, 0, stream>>>(attA, attB, PvT, lamm1,
                                                     adj, W_out, hsb);
  final_kernel<<<NN / 16, 32, 0, stream>>>(hsb, WlT, (float*)d_out);
}
